// TransformerBlock_18322330484922
// MI455X (gfx1250) — compile-verified, matrix operand order checked by reading, not run on hardware
//
#include <hip/hip_runtime.h>
#include <hip/hip_bf16.h>
#include <math.h>

// ---------------------------------------------------------------------------
// Types for CDNA5 WMMA (wave32): D(16x16 f32) = A(16x32 bf16) x B(32x16 bf16) + C
// ---------------------------------------------------------------------------
typedef __bf16 bf16x16 __attribute__((ext_vector_type(16)));
typedef float  f32x8   __attribute__((ext_vector_type(8)));
typedef unsigned int u32x4 __attribute__((ext_vector_type(4)));
typedef int  i32x4  __attribute__((ext_vector_type(4)));
typedef int  i32x8  __attribute__((ext_vector_type(8)));
typedef unsigned short ushort_t;

#define CC   512              // channels
#define BB   8
#define HH   48
#define WW   192
#define TT   (BB * HH * WW)   // 73728 tokens

#if defined(__has_builtin)
#if __has_builtin(__builtin_amdgcn_tensor_load_to_lds) && \
    __has_builtin(__builtin_amdgcn_s_wait_tensorcnt)
#define HAVE_TDM 1
#endif
#endif

__device__ __forceinline__ ushort_t f2bf(float f) {
    union { float f; unsigned u; } x; x.f = f;
    unsigned u = x.u + 0x7FFFu + ((x.u >> 16) & 1u);   // round-to-nearest-even
    return (ushort_t)(u >> 16);
}

#if defined(HAVE_TDM)
// Issue a TDM 2-D tile load: tileK x tileRows elements of 2 bytes, global row
// stride = rowStrideElems, packed row-major into LDS at lds_off.
// D# bitfields per CDNA5 ISA ch.8 (group0: count/lds/global/type; group1:
// data_size, tensor_dim0/1, tile_dim0/1, tensor_dim0_stride).
__device__ __forceinline__ void tdm_load_2d(unsigned lds_off, const void* gaddr,
                                            unsigned tileK, unsigned tileRows,
                                            unsigned rowStrideElems) {
    unsigned long long ga = (unsigned long long)gaddr;
    u32x4 g0;
    g0[0] = 1u;                                            // count=1 (user D#)
    g0[1] = lds_off;                                       // lds_addr [63:32]
    g0[2] = (unsigned)(ga & 0xFFFFFFFFu);                  // global_addr[31:0]
    g0[3] = (unsigned)((ga >> 32) & 0x01FFFFFFu) | 0x80000000u; // addr[56:32] | type=2
    i32x8 g1;
    g1[0] = (int)(1u << 16);                               // data_size=1 (2B)
    g1[1] = (int)((tileK & 0xFFFFu) << 16);                // tensor_dim0[15:0]
    g1[2] = (int)(((tileK >> 16) & 0xFFFFu) |              // tensor_dim0[31:16]
                  ((tileRows & 0xFFFFu) << 16));           // tensor_dim1[15:0]
    g1[3] = (int)(((tileRows >> 16) & 0xFFFFu) |           // tensor_dim1[31:16]
                  ((tileK & 0xFFFFu) << 16));              // tile_dim0
    g1[4] = (int)(tileRows & 0xFFFFu);                     // tile_dim1 (tile_dim2=0)
    g1[5] = (int)rowStrideElems;                           // tensor_dim0_stride[31:0]
    g1[6] = 0;                                             // stride[47:32] | dim1_stride lo
    g1[7] = 0;
    i32x4 z4 = {0, 0, 0, 0};
#if __clang_major__ >= 23
    i32x8 z8 = {0, 0, 0, 0, 0, 0, 0, 0};
    __builtin_amdgcn_tensor_load_to_lds(g0, g1, z4, z4, z8, 0);
#else
    __builtin_amdgcn_tensor_load_to_lds(g0, g1, z4, z4, 0);
#endif
}
#endif

// Load a 16x32 bf16 fragment from an LDS tile laid out [row][ld] starting at
// column kofs.  Per CDNA5 ISA 16-bit A/B layout: lane = (row%16) + 16*khalf,
// lane's 16 elements are K = khalf*8 + {0..7} and K = 16 + khalf*8 + {0..7}
// -> exactly two 16-byte LDS loads per lane.
__device__ __forceinline__ bf16x16 load_frag(const ushort_t* lds, int ld, int rowBase,
                                             int kofs, int lane) {
    int r  = rowBase + (lane & 15);
    int kh = (lane >> 4) << 3;                 // 0 or 8
    union { uint4 u[2]; bf16x16 v; } f;
    f.u[0] = *(const uint4*)(lds + r * ld + kofs + kh);
    f.u[1] = *(const uint4*)(lds + r * ld + kofs + 16 + kh);
    return f.v;
}

__device__ __forceinline__ f32x8 wmma_bf16(bf16x16 a, bf16x16 b, f32x8 c) {
    return __builtin_amdgcn_wmma_f32_16x16x32_bf16(false, a, false, b, (short)0, c,
                                                   false, false);
}

// ---------------------------------------------------------------------------
// Generic GEMM:  C[M,N] (op) = A[M,K](bf16, row-major) * B[N,K](bf16, row-major)
// OP 0: C = bf16(acc)
// OP 1: C = bf16(relu(acc + bias))
// OP 2: Cf32 += acc (+ bias if non-null)     (residual accumulate)
// Block: 256 threads = 8 waves; tile 128(M) x 128(N); K-stage 64.
// Each wave owns a 32(M) x 64(N) strip -> 16 WMMAs per stage between barriers.
// LDS staging via Tensor Data Mover (wave 0 posts two 2-D tile descriptors,
// s_wait_tensorcnt, barrier); falls back to vector loads if TDM unavailable.
// ---------------------------------------------------------------------------
template <int OP>
__global__ __launch_bounds__(256) void gemm_tn(const ushort_t* __restrict__ A,
                                               const ushort_t* __restrict__ B,
                                               void* __restrict__ Cp,
                                               const float* __restrict__ bias,
                                               int M, int N, int K) {
    __shared__ ushort_t lA[128 * 64];
    __shared__ ushort_t lB[128 * 64];
    const int tid  = threadIdx.x;
    const int wave = tid >> 5, lane = tid & 31;
    const int m0 = blockIdx.x * 128;
    const int n0 = blockIdx.y * 128;
    const int wm = ((wave >> 1) << 5);   // 0,32,64,96 within tile
    const int wn = ((wave & 1) << 6);    // 0,64

    f32x8 acc[2][4] = {};
    const int stages = K >> 6;
    for (int st = 0; st < stages; ++st) {
        const int k0 = st << 6;
#if defined(HAVE_TDM)
        if (wave == 0) {
            tdm_load_2d((unsigned)(size_t)lA, A + (size_t)m0 * K + k0, 64, 128, K);
            tdm_load_2d((unsigned)(size_t)lB, B + (size_t)n0 * K + k0, 64, 128, K);
            __builtin_amdgcn_s_wait_tensorcnt((short)0);
        }
#else
        {   // stage A and B tiles: 128 rows x 64 cols each, 4x b128 per thread
            const int r  = tid >> 1;
            const int hh = (tid & 1) << 5;
            const uint4* srcA = (const uint4*)(A + (size_t)(m0 + r) * K + k0 + hh);
            uint4* dstA = (uint4*)(lA + r * 64 + hh);
            dstA[0] = srcA[0]; dstA[1] = srcA[1]; dstA[2] = srcA[2]; dstA[3] = srcA[3];
            const uint4* srcB = (const uint4*)(B + (size_t)(n0 + r) * K + k0 + hh);
            uint4* dstB = (uint4*)(lB + r * 64 + hh);
            dstB[0] = srcB[0]; dstB[1] = srcB[1]; dstB[2] = srcB[2]; dstB[3] = srcB[3];
        }
#endif
        __syncthreads();
#pragma unroll
        for (int kk = 0; kk < 2; ++kk) {
            const int ko = kk << 5;      // 0, 32
            bf16x16 a0 = load_frag(lA, 64, wm,      ko, lane);
            bf16x16 a1 = load_frag(lA, 64, wm + 16, ko, lane);
            bf16x16 b0 = load_frag(lB, 64, wn,      ko, lane);
            bf16x16 b1 = load_frag(lB, 64, wn + 16, ko, lane);
            bf16x16 b2 = load_frag(lB, 64, wn + 32, ko, lane);
            bf16x16 b3 = load_frag(lB, 64, wn + 48, ko, lane);
            acc[0][0] = wmma_bf16(a0, b0, acc[0][0]);
            acc[0][1] = wmma_bf16(a0, b1, acc[0][1]);
            acc[0][2] = wmma_bf16(a0, b2, acc[0][2]);
            acc[0][3] = wmma_bf16(a0, b3, acc[0][3]);
            acc[1][0] = wmma_bf16(a1, b0, acc[1][0]);
            acc[1][1] = wmma_bf16(a1, b1, acc[1][1]);
            acc[1][2] = wmma_bf16(a1, b2, acc[1][2]);
            acc[1][3] = wmma_bf16(a1, b3, acc[1][3]);
        }
        __syncthreads();
    }
    // store: lane n = lane&15 ; row m = rr + 8*(lane>>4) per acc element rr
    const int ln = lane & 15, lh = lane >> 4;
    for (int i = 0; i < 2; ++i)
        for (int j = 0; j < 4; ++j) {
            const int bn = n0 + wn + j * 16 + ln;
            const float bv = (OP != 0 && bias != nullptr) ? bias[bn] : 0.f;
            for (int rr = 0; rr < 8; ++rr) {
                const int m = m0 + wm + i * 16 + rr + 8 * lh;
                float v = acc[i][j][rr];
                if (OP == 0) {
                    ((ushort_t*)Cp)[(size_t)m * N + bn] = f2bf(v);
                } else if (OP == 1) {
                    v += bv; v = v > 0.f ? v : 0.f;
                    ((ushort_t*)Cp)[(size_t)m * N + bn] = f2bf(v);
                } else {
                    ((float*)Cp)[(size_t)m * N + bn] += v + bv;
                }
            }
        }
}

// ---------------------------------------------------------------------------
// Batched attention logits: per sequence s, logits = scale * Q K^T.
// Sequence base token = (s/Sdiv)*Sslab + (s%Sdiv)*Srem ; tokens step strideTok.
// Block 128 thr (4 waves) computes one 64x64 tile: grid (numSeq, tq*nTiles+tk).
// ---------------------------------------------------------------------------
__global__ __launch_bounds__(128) void attn_logits(const ushort_t* __restrict__ Q,
                                                   const ushort_t* __restrict__ Kb,
                                                   float* __restrict__ logits,
                                                   int L, int nTiles, int Sdiv, int Sslab,
                                                   int Srem, int strideTok, float scale) {
    __shared__ ushort_t lQ[64 * 32];
    __shared__ ushort_t lK[64 * 32];
    const int s  = blockIdx.x;
    const int tq = blockIdx.y / nTiles, tk = blockIdx.y % nTiles;
    const size_t base = (size_t)(s / Sdiv) * Sslab + (size_t)(s % Sdiv) * Srem;
    const int tid = threadIdx.x, wave = tid >> 5, lane = tid & 31;

    f32x8 acc[4] = {};
    for (int k0 = 0; k0 < CC; k0 += 32) {
        const int r = tid >> 1, hh = (tid & 1) << 4;
        {
            uint4 z0{}, z1{};
            const int row = tq * 64 + r;
            if (row < L) {
                const uint4* src =
                    (const uint4*)(Q + (base + (size_t)row * strideTok) * CC + k0 + hh);
                z0 = src[0]; z1 = src[1];
            }
            uint4* d = (uint4*)(lQ + r * 32 + hh); d[0] = z0; d[1] = z1;
        }
        {
            uint4 z0{}, z1{};
            const int row = tk * 64 + r;
            if (row < L) {
                const uint4* src =
                    (const uint4*)(Kb + (base + (size_t)row * strideTok) * CC + k0 + hh);
                z0 = src[0]; z1 = src[1];
            }
            uint4* d = (uint4*)(lK + r * 32 + hh); d[0] = z0; d[1] = z1;
        }
        __syncthreads();
        bf16x16 aq = load_frag(lQ, 32, wave * 16, 0, lane);
        for (int j = 0; j < 4; ++j) {
            bf16x16 bk = load_frag(lK, 32, j * 16, 0, lane);
            acc[j] = wmma_bf16(aq, bk, acc[j]);
        }
        __syncthreads();
    }
    const int ln = lane & 15, lh = lane >> 4;
    for (int j = 0; j < 4; ++j) {
        const int n = tk * 64 + j * 16 + ln;
        for (int rr = 0; rr < 8; ++rr) {
            const int m = tq * 64 + wave * 16 + rr + 8 * lh;
            if (m < L && n < L)
                logits[((size_t)s * L + m) * L + n] = acc[j][rr] * scale;
        }
    }
}

// ---------------------------------------------------------------------------
// Row softmax: logits (rows of length L, f32) -> P (rows padded to Lpad, bf16)
// grid = numSeq * L, block = 128.
// ---------------------------------------------------------------------------
__global__ __launch_bounds__(128) void softmax_rows(const float* __restrict__ logits,
                                                    ushort_t* __restrict__ P,
                                                    int L, int Lpad) {
    const int row = blockIdx.x;
    const float* src = logits + (size_t)row * L;
    ushort_t* dst = P + (size_t)row * Lpad;
    __shared__ float red[128];
    const int tid = threadIdx.x;

    float mx = -1e30f;
    for (int i = tid; i < L; i += 128) mx = fmaxf(mx, src[i]);
    red[tid] = mx; __syncthreads();
    for (int o = 64; o; o >>= 1) { if (tid < o) red[tid] = fmaxf(red[tid], red[tid + o]); __syncthreads(); }
    mx = red[0]; __syncthreads();

    float sum = 0.f;
    for (int i = tid; i < L; i += 128) sum += __expf(src[i] - mx);
    red[tid] = sum; __syncthreads();
    for (int o = 64; o; o >>= 1) { if (tid < o) red[tid] += red[tid + o]; __syncthreads(); }
    const float inv = 1.0f / red[0];

    for (int i = tid; i < Lpad; i += 128)
        dst[i] = f2bf(i < L ? __expf(src[i] - mx) * inv : 0.f);
}

// ---------------------------------------------------------------------------
// Batched ctx = P(L x Lpad, bf16) @ V(L x C).  V staged transposed into LDS so
// the B fragment loader is the contiguous-K path.  Rows k >= L are clamped
// (their P entries are zero).  Block 128 thr -> 64(M) x 64(N) tile.
// grid = (numSeq, tm*8 + tn) with 8 = C/64 column tiles.
// ---------------------------------------------------------------------------
__global__ __launch_bounds__(128) void attn_ctx(const ushort_t* __restrict__ P,
                                                const ushort_t* __restrict__ V,
                                                ushort_t* __restrict__ ctx,
                                                int L, int Lpad, int Sdiv, int Sslab,
                                                int Srem, int strideTok) {
    __shared__ ushort_t lP[64 * 32];
    __shared__ ushort_t lV[64 * 32];
    const int s  = blockIdx.x;
    const int tm = blockIdx.y >> 3, tn = blockIdx.y & 7;
    const size_t base = (size_t)(s / Sdiv) * Sslab + (size_t)(s % Sdiv) * Srem;
    const int tid = threadIdx.x, wave = tid >> 5, lane = tid & 31;

    f32x8 acc[4] = {};
    for (int k0 = 0; k0 < Lpad; k0 += 32) {
        {   // P tile: rows m, cols k (contiguous)
            const int r = tid >> 1, hh = (tid & 1) << 4;
            uint4 z0{}, z1{};
            const int row = tm * 64 + r;
            if (row < L) {
                const uint4* src = (const uint4*)(P + ((size_t)s * L + row) * Lpad + k0 + hh);
                z0 = src[0]; z1 = src[1];
            }
            uint4* d = (uint4*)(lP + r * 32 + hh); d[0] = z0; d[1] = z1;
        }
        {   // V tile transposed: lV[n][k] ; vector global read, scalar LDS scatter
            const int kk = tid & 31;
            const int nb = (tid >> 5) << 4;            // 0,16,32,48
            int krow = k0 + kk; if (krow >= L) krow = L - 1;
            const ushort_t* src = V + (base + (size_t)krow * strideTok) * CC + tn * 64 + nb;
            ushort_t tmp[16];
            *(uint4*)tmp       = *(const uint4*)src;
            *(uint4*)(tmp + 8) = *(const uint4*)(src + 8);
            for (int i = 0; i < 16; ++i) lV[(nb + i) * 32 + kk] = tmp[i];
        }
        __syncthreads();
        bf16x16 ap = load_frag(lP, 32, wave * 16, 0, lane);
        for (int j = 0; j < 4; ++j) {
            bf16x16 bv = load_frag(lV, 32, j * 16, 0, lane);
            acc[j] = wmma_bf16(ap, bv, acc[j]);
        }
        __syncthreads();
    }
    const int ln = lane & 15, lh = lane >> 4;
    for (int j = 0; j < 4; ++j) {
        const int n = tn * 64 + j * 16 + ln;
        for (int rr = 0; rr < 8; ++rr) {
            const int m = tm * 64 + wave * 16 + rr + 8 * lh;
            if (m < L) {
                const size_t tok = base + (size_t)m * strideTok;
                ctx[tok * CC + n] = f2bf(acc[j][rr]);
            }
        }
    }
}

// ---------------------------------------------------------------------------
// LayerNorm (per token, C=512) -> bf16.  Block 256, two channels/thread.
// ---------------------------------------------------------------------------
__global__ __launch_bounds__(256) void ln_to_bf16(const float* __restrict__ X,
                                                  const float* __restrict__ g,
                                                  const float* __restrict__ b,
                                                  ushort_t* __restrict__ Y) {
    const int t = blockIdx.x, tid = threadIdx.x;
    const float* x = X + (size_t)t * CC;
    const float v0 = x[tid], v1 = x[tid + 256];
    __shared__ float rs[256], rq[256];
    rs[tid] = v0 + v1;
    rq[tid] = v0 * v0 + v1 * v1;
    __syncthreads();
    for (int o = 128; o; o >>= 1) {
        if (tid < o) { rs[tid] += rs[tid + o]; rq[tid] += rq[tid + o]; }
        __syncthreads();
    }
    const float mean = rs[0] * (1.0f / CC);
    const float var  = rq[0] * (1.0f / CC) - mean * mean;
    const float inv  = rsqrtf(var + 1e-5f);
    ushort_t* y = Y + (size_t)t * CC;
    y[tid]       = f2bf((v0 - mean) * inv * g[tid] + b[tid]);
    y[tid + 256] = f2bf((v1 - mean) * inv * g[tid + 256] + b[tid + 256]);
}

// ---------------------------------------------------------------------------
// Layout shuffles and weight conversion
// ---------------------------------------------------------------------------
__global__ void permute_in(const float* __restrict__ x, float* __restrict__ xt) {
    const size_t idx = (size_t)blockIdx.x * blockDim.x + threadIdx.x; // over T*C
    const int c = idx & (CC - 1);
    const size_t t = idx >> 9;
    const int w = t % WW, h = (t / WW) % HH, bb = t / (WW * HH);
    xt[idx] = x[(((size_t)bb * CC + c) * HH + h) * WW + w];
}

__global__ void permute_out(const float* __restrict__ xt, float* __restrict__ out) {
    const size_t idx = (size_t)blockIdx.x * blockDim.x + threadIdx.x; // over B*C*H*W
    const int w = idx % WW;
    const int h = (idx / WW) % HH;
    const int c = (idx / (WW * HH)) % CC;
    const int bb = idx / ((size_t)WW * HH * CC);
    out[idx] = xt[(((size_t)bb * HH + h) * WW + w) * CC + c];
}

__global__ void conv_w_bf16(const float* __restrict__ src, ushort_t* __restrict__ dst, int n) {
    const int i = blockIdx.x * blockDim.x + threadIdx.x;
    if (i < n) dst[i] = f2bf(src[i]);
}

// w_qkv rows interleave as d = 3*cout + j (torch reshape(...,C,3) semantics)
__global__ void deint_qkv(const float* __restrict__ w, ushort_t* __restrict__ wq,
                          ushort_t* __restrict__ wk, ushort_t* __restrict__ wv) {
    const int idx = blockIdx.x * blockDim.x + threadIdx.x;   // over 3*C*C
    const int c = idx & (CC - 1);
    const int d = idx >> 9;
    const int j = d % 3, cout = d / 3;
    const ushort_t bv = f2bf(w[idx]);
    const size_t o = (size_t)cout * CC + c;
    if (j == 0) wq[o] = bv; else if (j == 1) wk[o] = bv; else wv[o] = bv;
}

// ---------------------------------------------------------------------------
extern "C" void kernel_launch(void* const* d_in, const int* in_sizes, int n_in,
                              void* d_out, int out_size, void* d_ws, size_t ws_size,
                              hipStream_t stream) {
    const float* x      = (const float*)d_in[0];
    const float* w_qkv1 = (const float*)d_in[1];
    const float* w_out1 = (const float*)d_in[2];
    const float* w_qkv2 = (const float*)d_in[3];
    const float* w_out2 = (const float*)d_in[4];
    const float* g1 = (const float*)d_in[5];  const float* b1 = (const float*)d_in[6];
    const float* g2 = (const float*)d_in[7];  const float* b2 = (const float*)d_in[8];
    const float* g3 = (const float*)d_in[9];  const float* b3 = (const float*)d_in[10];
    const float* w_fc1 = (const float*)d_in[11]; const float* b_fc1 = (const float*)d_in[12];
    const float* w_fc2 = (const float*)d_in[13]; const float* b_fc2 = (const float*)d_in[14];
    float* out = (float*)d_out;

    // ---- workspace carve-up (h1 aliases the dead q..P attention region) ----
    char* ws = (char*)d_ws;
    size_t off = 0;
    auto carve = [&](size_t bytes) { size_t r = off; off = (off + bytes + 255) & ~(size_t)255; return r; };
    const size_t o_xt   = carve((size_t)TT * CC * 4);       // f32 activations
    const size_t o_yh   = carve((size_t)TT * CC * 2);       // bf16 LN output
    const size_t o_q    = carve((size_t)TT * CC * 2);
    const size_t o_k    = carve((size_t)TT * CC * 2);
    const size_t o_v    = carve((size_t)TT * CC * 2);
    const size_t o_lg   = carve((size_t)384 * 192 * 192 * 4);   // max logits (attn2)
    const size_t o_p    = carve((size_t)384 * 192 * 192 * 2);   // max P (attn2)
    const size_t o_ctx  = carve((size_t)TT * CC * 2);
    const size_t o_wq1  = carve((size_t)CC * CC * 2);
    const size_t o_wk1  = carve((size_t)CC * CC * 2);
    const size_t o_wv1  = carve((size_t)CC * CC * 2);
    const size_t o_wo1  = carve((size_t)CC * CC * 2);
    const size_t o_wq2  = carve((size_t)CC * CC * 2);
    const size_t o_wk2  = carve((size_t)CC * CC * 2);
    const size_t o_wv2  = carve((size_t)CC * CC * 2);
    const size_t o_wo2  = carve((size_t)CC * CC * 2);
    const size_t o_wf1  = carve((size_t)4 * CC * CC * 2);
    const size_t o_wf2  = carve((size_t)4 * CC * CC * 2);
    (void)ws_size; (void)n_in; (void)in_sizes; (void)out_size;

    float*    xt  = (float*)(ws + o_xt);
    ushort_t* yh  = (ushort_t*)(ws + o_yh);
    ushort_t* qb  = (ushort_t*)(ws + o_q);
    ushort_t* kb  = (ushort_t*)(ws + o_k);
    ushort_t* vb  = (ushort_t*)(ws + o_v);
    float*    lg  = (float*)(ws + o_lg);
    ushort_t* pb  = (ushort_t*)(ws + o_p);
    ushort_t* ctx = (ushort_t*)(ws + o_ctx);
    ushort_t* h1  = (ushort_t*)(ws + o_q);   // alias: 302 MB fits in q..P region
    ushort_t* wq1 = (ushort_t*)(ws + o_wq1); ushort_t* wk1 = (ushort_t*)(ws + o_wk1);
    ushort_t* wv1 = (ushort_t*)(ws + o_wv1); ushort_t* wo1 = (ushort_t*)(ws + o_wo1);
    ushort_t* wq2 = (ushort_t*)(ws + o_wq2); ushort_t* wk2 = (ushort_t*)(ws + o_wk2);
    ushort_t* wv2 = (ushort_t*)(ws + o_wv2); ushort_t* wo2 = (ushort_t*)(ws + o_wo2);
    ushort_t* wf1 = (ushort_t*)(ws + o_wf1); ushort_t* wf2 = (ushort_t*)(ws + o_wf2);

    const float scale = 1.0f / sqrtf((float)CC);

    // ---- weight conversion (deterministic, every call) ----
    deint_qkv<<<(3 * CC * CC) / 256, 256, 0, stream>>>(w_qkv1, wq1, wk1, wv1);
    deint_qkv<<<(3 * CC * CC) / 256, 256, 0, stream>>>(w_qkv2, wq2, wk2, wv2);
    conv_w_bf16<<<(CC * CC) / 256, 256, 0, stream>>>(w_out1, wo1, CC * CC);
    conv_w_bf16<<<(CC * CC) / 256, 256, 0, stream>>>(w_out2, wo2, CC * CC);
    conv_w_bf16<<<(4 * CC * CC) / 256, 256, 0, stream>>>(w_fc1, wf1, 4 * CC * CC);
    conv_w_bf16<<<(4 * CC * CC) / 256, 256, 0, stream>>>(w_fc2, wf2, 4 * CC * CC);

    // ---- stage activations token-major (B,H,W,C) f32 ----
    permute_in<<<(TT * CC) / 256, 256, 0, stream>>>(x, xt);

    const dim3 gemmBlk(256);
    const dim3 gGemm512(TT / 128, CC / 128);          // 576 x 4
    const dim3 gGemm2048(TT / 128, (4 * CC) / 128);   // 576 x 16

    // ================= attention 1 (along H: stride W, 1536 sequences) ======
    ln_to_bf16<<<TT, 256, 0, stream>>>(xt, g1, b1, yh);
    gemm_tn<0><<<gGemm512, gemmBlk, 0, stream>>>(yh, wq1, qb, nullptr, TT, CC, CC);
    gemm_tn<0><<<gGemm512, gemmBlk, 0, stream>>>(yh, wk1, kb, nullptr, TT, CC, CC);
    gemm_tn<0><<<gGemm512, gemmBlk, 0, stream>>>(yh, wv1, vb, nullptr, TT, CC, CC);
    // seq s -> base = (s/W)*(H*W) + (s%W); L=48 (one 64-tile), Lpad=64
    attn_logits<<<dim3(BB * WW, 1), 128, 0, stream>>>(qb, kb, lg, HH, 1, WW, HH * WW, 1, WW, scale);
    softmax_rows<<<BB * WW * HH, 128, 0, stream>>>(lg, pb, HH, 64);
    attn_ctx<<<dim3(BB * WW, 1 * 8), 128, 0, stream>>>(pb, vb, ctx, HH, 64, WW, HH * WW, 1, WW);
    gemm_tn<2><<<gGemm512, gemmBlk, 0, stream>>>(ctx, wo1, xt, nullptr, TT, CC, CC);

    // ================= attention 2 (along W: stride 1, 384 sequences) =======
    ln_to_bf16<<<TT, 256, 0, stream>>>(xt, g2, b2, yh);
    gemm_tn<0><<<gGemm512, gemmBlk, 0, stream>>>(yh, wq2, qb, nullptr, TT, CC, CC);
    gemm_tn<0><<<gGemm512, gemmBlk, 0, stream>>>(yh, wk2, kb, nullptr, TT, CC, CC);
    gemm_tn<0><<<gGemm512, gemmBlk, 0, stream>>>(yh, wv2, vb, nullptr, TT, CC, CC);
    // seq s -> base = s*W; L=192 (3x3 64-tiles), Lpad=192
    attn_logits<<<dim3(BB * HH, 9), 128, 0, stream>>>(qb, kb, lg, WW, 3, 1, WW, 0, 1, scale);
    softmax_rows<<<BB * HH * WW, 128, 0, stream>>>(lg, pb, WW, WW);
    attn_ctx<<<dim3(BB * HH, 3 * 8), 128, 0, stream>>>(pb, vb, ctx, WW, WW, 1, WW, 0, 1);
    gemm_tn<2><<<gGemm512, gemmBlk, 0, stream>>>(ctx, wo2, xt, nullptr, TT, CC, CC);

    // ================= MLP ==================================================
    ln_to_bf16<<<TT, 256, 0, stream>>>(xt, g3, b3, yh);
    gemm_tn<1><<<gGemm2048, gemmBlk, 0, stream>>>(yh, wf1, h1, b_fc1, TT, 4 * CC, CC);
    gemm_tn<2><<<gGemm512, gemmBlk, 0, stream>>>(h1, wf2, xt, b_fc2, TT, CC, 4 * CC);

    // ---- back to (B,C,H,W) ----
    permute_out<<<(TT * CC) / 256, 256, 0, stream>>>(xt, out);
}
